// Our1_18580028522784
// MI455X (gfx1250) — matrix operand order repaired and checked by reading, not compiled, not run on hardware
//
#include <hip/hip_runtime.h>
#include <hip/hip_bf16.h>

// ---------------- problem constants (match reference) ----------------
#define N_HEADN 50000
#define N_TAILN 50000
#define NEDGE   800000
#define IN_F    256
#define OUT_F   64
#define NHEAD   4
#define OH      256   // OUT_F * NHEAD
#define EFDIM   64
#define NTYPE   8
#define NEG_SLOPE 0.2f

typedef __attribute__((ext_vector_type(16))) __bf16 v16bf;
typedef __attribute__((ext_vector_type(2)))  __bf16 v2bf;
typedef __attribute__((ext_vector_type(16))) float  v16f;
typedef __attribute__((ext_vector_type(2)))  float  v2f;
typedef __attribute__((ext_vector_type(8)))  float  v8f;

// packed f32 -> bf16 conversions: vector fptrunc lets the backend pick
// v_cvt_pk_bf16_f32 when the target has it.
__device__ __forceinline__ v2bf pk2(float x, float y) {
#if __has_builtin(__builtin_amdgcn_cvt_pk_bf16_f32)
  return __builtin_amdgcn_cvt_pk_bf16_f32(x, y);
#else
  v2f t; t[0] = x; t[1] = y;
  return __builtin_convertvector(t, v2bf);
#endif
}

// order-preserving float <-> uint encoding for atomicMax-based segment max
__device__ __forceinline__ unsigned encf(float f) {
  unsigned u = __float_as_uint(f);
  return (u & 0x80000000u) ? ~u : (u | 0x80000000u);
}
__device__ __forceinline__ float decf(unsigned u) {
  unsigned v = (u & 0x80000000u) ? (u & 0x7FFFFFFFu) : ~u;
  return __uint_as_float(v);
}
__device__ __forceinline__ float lrelu(float x) { return x > 0.f ? x : NEG_SLOPE * x; }

// ---------------------------------------------------------------------
// Fused GEMM + attention row-dot:
//   Yfull[rows, 256] = X[rows, 256] @ Wm[256, 256]   (stored only if Y != null)
//   hout[row, h]     = dot(avec[h, :], Yfull[row, h*64 : (h+1)*64])
// block = 256 threads (8 waves, wave32). grid = (ceil(rows/128), NHEAD):
// blockIdx.y selects the 64-column chunk == attention head.
// W chunk staged transposed ([col][k]) in 32KB LDS as bf16;
// each wave computes a 16x64 tile with 4x v_wmma_f32_16x16x32_bf16 per K step.
// ---------------------------------------------------------------------
__global__ __launch_bounds__(256)
void gemm256_wmma(const float* __restrict__ X, int nrows,
                  const float* __restrict__ Wm, float* __restrict__ Y,
                  const float* __restrict__ avec, float* __restrict__ hout) {
  __shared__ __bf16 sW[64 * IN_F];               // 32 KB, [col][k]
  const int hq = blockIdx.y;                     // attention head index
  const int colBase = hq * 64;
  const int tid = threadIdx.x;

  // stage W chunk: pack 2 consecutive k per thread-iteration
  for (int i = tid; i < 64 * (IN_F / 2); i += 256) {
    int c = i & 63, k = (i >> 6) * 2;
    float w0 = Wm[k * OH + colBase + c];
    float w1 = Wm[(k + 1) * OH + colBase + c];
    *(v2bf*)&sW[c * IN_F + k] = pk2(w0, w1);
  }
  __syncthreads();

  const int wave = tid >> 5;                     // wave32
  const int lane = tid & 31;
  const int rowBase = blockIdx.x * 128 + wave * 16;
  const int mrow = lane & 15;
  const int hi   = lane >> 4;                    // half-wave select

  int arow = rowBase + mrow;
  if (arow >= nrows) arow = nrows - 1;           // clamp loads; stores guarded
  const float* xrow = X + (size_t)arow * IN_F;

  v8f acc[4] = {};

#pragma unroll 2
  for (int kt = 0; kt < IN_F / 32; ++kt) {
    // branchless prefetch of the next K chunk (wraps on the last iteration)
    __builtin_prefetch(xrow + ((kt + 1) & 7) * 32 + hi * 8, 0, 1);

    // A fragment (16-bit A layout): lanes 0-15 hold K {0..7,16..23}, lanes
    // 16-31 hold K {8..15,24..31} of this 32-wide K step.
    const int k0 = kt * 32 + hi * 8;
    float4 a0 = *(const float4*)(xrow + k0);
    float4 a1 = *(const float4*)(xrow + k0 + 4);
    float4 a2 = *(const float4*)(xrow + k0 + 16);
    float4 a3 = *(const float4*)(xrow + k0 + 20);
    v16f xa;
    xa[0]=a0.x;  xa[1]=a0.y;  xa[2]=a0.z;  xa[3]=a0.w;
    xa[4]=a1.x;  xa[5]=a1.y;  xa[6]=a1.z;  xa[7]=a1.w;
    xa[8]=a2.x;  xa[9]=a2.y;  xa[10]=a2.z; xa[11]=a2.w;
    xa[12]=a3.x; xa[13]=a3.y; xa[14]=a3.z; xa[15]=a3.w;
    v16bf af = __builtin_convertvector(xa, v16bf);   // packed cvt if available

    // load all B fragments (8x ds_load_b128), then 4 back-to-back WMMAs
    v16bf bf[4];
#pragma unroll
    for (int nt = 0; nt < 4; ++nt) {
      typedef __attribute__((ext_vector_type(8))) __bf16 v8bf_t;
      const __bf16* bp = &sW[(nt * 16 + mrow) * IN_F + kt * 32 + hi * 8];
      v8bf_t lo = *(const v8bf_t*)bp;
      v8bf_t hv = *(const v8bf_t*)(bp + 16);
      bf[nt] = __builtin_shufflevector(lo, hv, 0,1,2,3,4,5,6,7,8,9,10,11,12,13,14,15);
    }
#pragma unroll
    for (int nt = 0; nt < 4; ++nt)
      acc[nt] = __builtin_amdgcn_wmma_f32_16x16x32_bf16(
          false, af, false, bf[nt], (short)0, acc[nt], false, false);
  }

  // optional store of the full projection (needed only for tail features)
  // C/D layout: VGPR r -> M = r + 8*hi, N = lane&15 within each 16-wide tile
  if (Y) {
    const int col0 = colBase + mrow;
    float* yb = Y + (size_t)(rowBase + hi * 8) * OH + col0;
    if (rowBase + 16 <= nrows) {                 // uniform fast path: no guards
#pragma unroll
      for (int nt = 0; nt < 4; ++nt)
#pragma unroll
        for (int r = 0; r < 8; ++r)
          yb[(size_t)r * OH + nt * 16] = acc[nt][r];
    } else {
#pragma unroll
      for (int nt = 0; nt < 4; ++nt)
#pragma unroll
        for (int r = 0; r < 8; ++r)
          if (rowBase + hi * 8 + r < nrows)
            yb[(size_t)r * OH + nt * 16] = acc[nt][r];
    }
  }

  // fused attention row-dot epilogue: hout[row, hq] = sum_c avec[hq,c]*Y[row,c']
  float ca[4];
#pragma unroll
  for (int nt = 0; nt < 4; ++nt) ca[nt] = avec[colBase + nt * 16 + mrow];
#pragma unroll
  for (int r = 0; r < 8; ++r) {
    float s = fmaf(ca[0], acc[0][r],
              fmaf(ca[1], acc[1][r],
              fmaf(ca[2], acc[2][r], ca[3] * acc[3][r])));
    // reduce across the 16 lanes of this half-wave (same M row)
    s += __shfl_xor(s, 1, 32);
    s += __shfl_xor(s, 2, 32);
    s += __shfl_xor(s, 4, 32);
    s += __shfl_xor(s, 8, 32);
    int row = rowBase + hi * 8 + r;
    if (mrow == 0 && row < nrows) hout[row * NHEAD + hq] = s;
  }
}

// h_e[t,h] = sum_ef a_e[h,ef] * (edge_emb @ W_e)[t, h*64+ef]   (tiny)
__global__ void he_kernel(const float* __restrict__ emb, const float* __restrict__ We,
                          const float* __restrict__ ae, float* __restrict__ he) {
  int tid = threadIdx.x;
  if (tid >= NTYPE * NHEAD) return;
  int t = tid >> 2, h = tid & 3;
  float s = 0.f;
  for (int ef = 0; ef < EFDIM; ++ef) {
    float ev = 0.f;
    for (int j = 0; j < EFDIM; ++j)
      ev = fmaf(emb[t * EFDIM + j], We[j * (EFDIM * NHEAD) + h * EFDIM + ef], ev);
    s = fmaf(ae[h * EFDIM + ef], ev, s);
  }
  he[tid] = s;
}

__global__ void att_max_kernel(const int* __restrict__ heads, const int* __restrict__ tails,
                               const int* __restrict__ types,
                               const float* __restrict__ hl, const float* __restrict__ hr,
                               const float* __restrict__ he, unsigned* __restrict__ m) {
  int e = blockIdx.x * blockDim.x + threadIdx.x;
  if (e >= NEDGE) return;
  int hd = heads[e], tl = tails[e], ty = types[e];
#pragma unroll
  for (int h = 0; h < NHEAD; ++h) {
    float att = lrelu(hl[hd * NHEAD + h] + hr[tl * NHEAD + h] + he[ty * NHEAD + h]);
    atomicMax(&m[hd * NHEAD + h], encf(att));
  }
}

__global__ void att_den_kernel(const int* __restrict__ heads, const int* __restrict__ tails,
                               const int* __restrict__ types,
                               const float* __restrict__ hl, const float* __restrict__ hr,
                               const float* __restrict__ he, const unsigned* __restrict__ m,
                               float* __restrict__ den) {
  int e = blockIdx.x * blockDim.x + threadIdx.x;
  if (e >= NEDGE) return;
  int hd = heads[e], tl = tails[e], ty = types[e];
#pragma unroll
  for (int h = 0; h < NHEAD; ++h) {
    float att = lrelu(hl[hd * NHEAD + h] + hr[tl * NHEAD + h] + he[ty * NHEAD + h]);
    atomicAdd(&den[hd * NHEAD + h], __expf(att - decf(m[hd * NHEAD + h])));
  }
}

// one 256-thread block per edge: thread t -> (h = t>>6, o = t&63)
__global__ __launch_bounds__(256)
void scatter_kernel(const int* __restrict__ heads, const int* __restrict__ tails,
                    const int* __restrict__ types,
                    const float* __restrict__ hl, const float* __restrict__ hr,
                    const float* __restrict__ he, const unsigned* __restrict__ m,
                    const float* __restrict__ den, const float* __restrict__ htail,
                    float* __restrict__ out) {
  int e = blockIdx.x;
  int t = threadIdx.x;
  int h = t >> 6;
  int hd = heads[e], tl = tails[e], ty = types[e];
  float att = lrelu(hl[hd * NHEAD + h] + hr[tl * NHEAD + h] + he[ty * NHEAD + h]);
  float w = __expf(att - decf(m[hd * NHEAD + h])) / den[hd * NHEAD + h];
  atomicAdd(&out[(size_t)hd * OH + t], w * htail[(size_t)tl * OH + t]);
}

__global__ void elu_kernel(float* __restrict__ out, int n) {
  int i = blockIdx.x * blockDim.x + threadIdx.x;
  if (i >= n) return;
  float x = out[i];
  out[i] = x > 0.f ? x : (__expf(x) - 1.f);
}

__global__ void fill_f32(float* __restrict__ p, float v, int n) {
  int i = blockIdx.x * blockDim.x + threadIdx.x;
  if (i < n) p[i] = v;
}
__global__ void fill_u32(unsigned* __restrict__ p, unsigned v, int n) {
  int i = blockIdx.x * blockDim.x + threadIdx.x;
  if (i < n) p[i] = v;
}

extern "C" void kernel_launch(void* const* d_in, const int* in_sizes, int n_in,
                              void* d_out, int out_size, void* d_ws, size_t ws_size,
                              hipStream_t stream) {
  (void)in_sizes; (void)n_in; (void)out_size; (void)ws_size;
  const float* head_feat = (const float*)d_in[0];
  const float* tail_feat = (const float*)d_in[1];
  const float* Wm        = (const float*)d_in[2];
  const float* We        = (const float*)d_in[3];
  const float* emb       = (const float*)d_in[4];
  const float* a_l       = (const float*)d_in[5];
  const float* a_r       = (const float*)d_in[6];
  const float* a_e       = (const float*)d_in[7];
  const int*   el        = (const int*)d_in[8];
  const int*   types     = (const int*)d_in[9];
  const int* heads = el;
  const int* tails = el + NEDGE;
  float* out = (float*)d_out;

  // workspace layout (~55 MB); h_head never materialized
  float* ws = (float*)d_ws;
  float*    h_tail = ws;                                  // 50000*256
  float*    h_l    = h_tail + (size_t)N_TAILN * OH;       // 50000*4
  float*    h_r    = h_l + N_HEADN * NHEAD;               // 50000*4
  float*    h_e    = h_r + N_TAILN * NHEAD;               // 8*4
  unsigned* m      = (unsigned*)(h_e + NTYPE * NHEAD);    // 50000*4
  float*    den    = (float*)(m + N_HEADN * NHEAD);       // 50000*4

  const int OUT_N = N_HEADN * OH;
  const int SEG_N = N_HEADN * NHEAD;
  fill_f32<<<(OUT_N + 255) / 256, 256, 0, stream>>>(out, 0.f, OUT_N);
  fill_f32<<<(SEG_N + 255) / 256, 256, 0, stream>>>(den, 0.f, SEG_N);
  fill_u32<<<(SEG_N + 255) / 256, 256, 0, stream>>>(m, 0u, SEG_N);

  dim3 ggrid((N_HEADN + 127) / 128, NHEAD);
  // head projection: only h_l is needed downstream -> no Y store
  gemm256_wmma<<<ggrid, 256, 0, stream>>>(head_feat, N_HEADN, Wm, nullptr, a_l, h_l);
  // tail projection: h_tail needed for messages, h_r fused
  gemm256_wmma<<<ggrid, 256, 0, stream>>>(tail_feat, N_TAILN, Wm, h_tail, a_r, h_r);
  he_kernel<<<1, 32, 0, stream>>>(emb, We, a_e, h_e);
  att_max_kernel<<<(NEDGE + 255) / 256, 256, 0, stream>>>(heads, tails, types, h_l, h_r, h_e, m);
  att_den_kernel<<<(NEDGE + 255) / 256, 256, 0, stream>>>(heads, tails, types, h_l, h_r, h_e, m, den);
  scatter_kernel<<<NEDGE, 256, 0, stream>>>(heads, tails, types, h_l, h_r, h_e, m, den, h_tail, out);
  elu_kernel<<<(OUT_N + 255) / 256, 256, 0, stream>>>(out, OUT_N);
}
